// GPTOneLayer_60739427500245
// MI455X (gfx1250) — compile-verified, matrix operand order checked by reading, not run on hardware
//
#include <hip/hip_runtime.h>
#include <hip/hip_bf16.h>

// ---------------------------------------------------------------------------
// GPT one-layer forward on MI455X (gfx1250, wave32, WMMA).
// GEMMs: v_wmma_f32_16x16x32_bf16 fed from LDS, staged with double-buffered
// global_load_async_to_lds_b128 (ASYNCcnt) -- the CDNA5 async copy path.
// ---------------------------------------------------------------------------

#define S_LEN 2048
#define D_DIM 2048
#define H_HEADS 8
#define DH_DIM 256         // D / H
#define V_VOCAB 50257
#define DF_DIM 8192        // 4*D
#define NPAD_V 50304       // V padded up to multiple of 128 (393 * 128)
#define LN_EPS 1e-5f

#define BM 128
#define BN 128
#define BK 32
#define LDS_STRIDE 40      // 32 + 8 pad elements (80B = 20 banks; gcd(20,64)=4
                           // -> each 16-lane half of a b128 read covers all 64
                           // banks exactly once: conflict-free)

typedef __attribute__((ext_vector_type(16))) __bf16 v16bf;
typedef __attribute__((ext_vector_type(8)))  __bf16 v8bf;
typedef __attribute__((ext_vector_type(8)))  float  v8f;

// ---------------------------------------------------------------------------
// Async DMA: 16B global -> LDS, tracked by ASYNCcnt (no VGPR round trip).
// Generic pointers to __shared__ carry the LDS byte offset in their low 32
// bits (flat addressing), which is exactly what the
// instruction's LDS-address VGPR wants.
// ---------------------------------------------------------------------------
__device__ __forceinline__ void async_ld16(const __bf16* g, __bf16* l) {
  unsigned loff = (unsigned)(unsigned long long)l;
  asm volatile("global_load_async_to_lds_b128 %0, %1, off"
               :: "v"(loff), "v"(g) : "memory");
}

// ---------------------------------------------------------------------------
// Fragment loader for NT GEMM operands out of an LDS tile (row-major,
// LDS_STRIDE elements per row).  Per CDNA5 ISA 16-bit A/B layout:
//   lanes 0-15  hold row = l16, K chunks {0..7, 16..23}
//   lanes 16-31 hold row = l16, K chunks {8..15, 24..31}
// i.e. per lane: two contiguous 8-element (16B) chunks at koff and koff+16.
// ---------------------------------------------------------------------------
__device__ __forceinline__ v16bf load_frag_k32(const __bf16* p) {
  union { v16bf v; v8bf h[2]; } u;
  u.h[0] = *reinterpret_cast<const v8bf*>(p);
  u.h[1] = *reinterpret_cast<const v8bf*>(p + 16);
  return u.v;
}

// ---------------------------------------------------------------------------
// Tiled WMMA GEMM:  C[M,N] = scale * (A[M,K] x B[N,K]^T) + bias[N], epilogue.
// Block tile 128x128 (256 thr = 8 waves), wave tile 64(M) x 32(N);
// per K-step of 32: LDS double-buffer staged by 4 async b128 loads/thread,
// then 4 A-frags + 2 B-frags -> 8 WMMAs per wave.
// M and grid-covered N must be multiples of 128; K multiple of 32.
// Stores guarded by n < Nvalid.  epi: 0 = none, 1 = exact erf-GELU.
// ---------------------------------------------------------------------------
__global__ __launch_bounds__(256)
void wmma_gemm_nt(const __bf16* __restrict__ A, int lda,
                  const __bf16* __restrict__ B, int ldb,
                  const float*  __restrict__ bias,
                  float scale,
                  float*  __restrict__ Cf,    // fp32 output (nullable)
                  __bf16* __restrict__ Cb,    // bf16 output (nullable)
                  long ldc,
                  int Nvalid, int K, int epi)
{
  __shared__ __bf16 sA[2][BM * LDS_STRIDE];
  __shared__ __bf16 sB[2][BN * LDS_STRIDE];

  const int tid  = threadIdx.x;
  const int lane = tid & 31;
  const int wave = tid >> 5;
  const int half = lane >> 4;
  const int l16  = lane & 15;
  const int wm   = wave >> 2;                 // 0..1
  const int wn   = wave & 3;                  // 0..3
  const int bm0  = blockIdx.y * BM;
  const int bn0  = blockIdx.x * BN;

  // Staging: tile = 128 rows x 4 chunks of 16B; thread t does chunks t, t+256.
  const int srow = tid >> 2;                  // 0..63
  const int scc  = (tid & 3) * 8;             // K element offset of 16B chunk
  const __bf16* gA0 = A + (size_t)(bm0 + srow)      * (size_t)lda + scc;
  const __bf16* gA1 = A + (size_t)(bm0 + srow + 64) * (size_t)lda + scc;
  const __bf16* gB0 = B + (size_t)(bn0 + srow)      * (size_t)ldb + scc;
  const __bf16* gB1 = B + (size_t)(bn0 + srow + 64) * (size_t)ldb + scc;
  const int l0 = srow * LDS_STRIDE + scc;
  const int l1 = (srow + 64) * LDS_STRIDE + scc;

  v8f acc[4][2];
#pragma unroll
  for (int mi = 0; mi < 4; ++mi)
#pragma unroll
    for (int ni = 0; ni < 2; ++ni)
      acc[mi][ni] = v8f{0.f, 0.f, 0.f, 0.f, 0.f, 0.f, 0.f, 0.f};

  const int koff = half ? 8 : 0;
  const int nsteps = K / BK;

  // prologue: stage buffer 0
  async_ld16(gA0, &sA[0][l0]);
  async_ld16(gA1, &sA[0][l1]);
  async_ld16(gB0, &sB[0][l0]);
  async_ld16(gB1, &sB[0][l1]);

  for (int s = 0; s < nsteps; ++s) {
    const int cur = s & 1;
    const bool pf = (s + 1 < nsteps);
    if (pf) {
      const int kb = (s + 1) * BK;
      const int nxt = cur ^ 1;
      async_ld16(gA0 + kb, &sA[nxt][l0]);
      async_ld16(gA1 + kb, &sA[nxt][l1]);
      async_ld16(gB0 + kb, &sB[nxt][l0]);
      async_ld16(gB1 + kb, &sB[nxt][l1]);
      // current buffer's 4 loads done; the 4 just-issued may remain in flight
      asm volatile("s_wait_asynccnt 0x4" ::: "memory");
    } else {
      asm volatile("s_wait_asynccnt 0x0" ::: "memory");
    }
    __syncthreads();

    const __bf16* pa = &sA[cur][(wm * 64 + l16) * LDS_STRIDE + koff];
    const __bf16* pb = &sB[cur][(wn * 32 + l16) * LDS_STRIDE + koff];
    v16bf af[4], bfr[2];
#pragma unroll
    for (int mi = 0; mi < 4; ++mi)
      af[mi] = load_frag_k32(pa + mi * 16 * LDS_STRIDE);
#pragma unroll
    for (int ni = 0; ni < 2; ++ni)
      bfr[ni] = load_frag_k32(pb + ni * 16 * LDS_STRIDE);

#pragma unroll
    for (int mi = 0; mi < 4; ++mi)
#pragma unroll
      for (int ni = 0; ni < 2; ++ni)
        acc[mi][ni] = __builtin_amdgcn_wmma_f32_16x16x32_bf16(
            false, af[mi], false, bfr[ni],
            (short)0, acc[mi][ni], false, false);

    __syncthreads();  // everyone done reading cur before it is overwritten
  }

  // Epilogue.  C/D layout: VGPR r -> row m = r + 8*half, col n = l16.
#pragma unroll
  for (int mi = 0; mi < 4; ++mi) {
#pragma unroll
    for (int ni = 0; ni < 2; ++ni) {
      const int n = bn0 + wn * 32 + ni * 16 + l16;
      const float bsv = bias ? bias[n] : 0.f;
#pragma unroll
      for (int r = 0; r < 8; ++r) {
        const int m = bm0 + wm * 64 + mi * 16 + r + 8 * half;
        float v = acc[mi][ni][r] * scale + bsv;
        if (epi == 1) v = 0.5f * v * (1.0f + erff(v * 0.70710678118654752f));
        if (n < Nvalid) {
          const size_t idx = (size_t)m * (size_t)ldc + (size_t)n;
          if (Cf) Cf[idx] = v;
          if (Cb) Cb[idx] = (__bf16)v;
        }
      }
    }
  }
}

// ---------------------------------------------------------------------------
// Elementwise helpers
// ---------------------------------------------------------------------------
__global__ void cvt_f32_to_bf16(const float* __restrict__ in,
                                __bf16* __restrict__ out, long n) {
  long i = (long)blockIdx.x * blockDim.x + threadIdx.x;
  if (i < n) out[i] = (__bf16)in[i];
}

__global__ void fill_bf16_zero(__bf16* __restrict__ out, long n) {
  long i = (long)blockIdx.x * blockDim.x + threadIdx.x;
  if (i < n) out[i] = (__bf16)0.f;
}

// x[s,d] = Wemb[token_ids[s], d] + Wpos[s, d]   (fp32 + bf16 copies)
__global__ void embed_kernel(const int* __restrict__ tok,
                             const float* __restrict__ Wemb,
                             const float* __restrict__ Wpos,
                             float* __restrict__ xf,
                             __bf16* __restrict__ xb) {
  long i = (long)blockIdx.x * blockDim.x + threadIdx.x;
  if (i >= (long)S_LEN * D_DIM) return;
  int s = (int)(i / D_DIM);
  int d = (int)(i % D_DIM);
  float v = Wemb[(size_t)tok[s] * D_DIM + d] + Wpos[(size_t)s * D_DIM + d];
  xf[i] = v;
  xb[i] = (__bf16)v;
}

// vT[d, t] = v[t, h*DH + d]   (per-head transpose for NT attn GEMM)
__global__ void transpose_head(const __bf16* __restrict__ v,
                               __bf16* __restrict__ vT, int h) {
  long i = (long)blockIdx.x * blockDim.x + threadIdx.x;
  if (i >= (long)DH_DIM * S_LEN) return;
  int d = (int)(i / S_LEN);
  int t = (int)(i % S_LEN);
  vT[(size_t)d * S_LEN + t] = v[(size_t)t * D_DIM + (size_t)h * DH_DIM + d];
}

// Causal softmax, one row per block; masked tail written as exact bf16 zero.
__global__ __launch_bounds__(256)
void softmax_causal(const float* __restrict__ scores,
                    __bf16* __restrict__ probs) {
  __shared__ float red[256];
  const int row = blockIdx.x;
  const int tid = threadIdx.x;
  const int L = row + 1;
  const float* sr = scores + (size_t)row * S_LEN;
  __bf16* pr = probs + (size_t)row * S_LEN;

  float mx = -3.4e38f;
  for (int j = tid; j < L; j += 256) mx = fmaxf(mx, sr[j]);
  red[tid] = mx; __syncthreads();
  for (int st = 128; st > 0; st >>= 1) {
    if (tid < st) red[tid] = fmaxf(red[tid], red[tid + st]);
    __syncthreads();
  }
  mx = red[0]; __syncthreads();

  float sum = 0.f;
  for (int j = tid; j < L; j += 256) sum += expf(sr[j] - mx);
  red[tid] = sum; __syncthreads();
  for (int st = 128; st > 0; st >>= 1) {
    if (tid < st) red[tid] += red[tid + st];
    __syncthreads();
  }
  const float inv = 1.f / red[0];

  for (int j = tid; j < S_LEN; j += 256)
    pr[j] = (j < L) ? (__bf16)(expf(sr[j] - mx) * inv) : (__bf16)0.f;
}

// hn = LayerNorm(x + attn) * g + b  -> bf16 (one row per block)
__global__ __launch_bounds__(256)
void residual_layernorm(const float* __restrict__ x,
                        const float* __restrict__ attn,
                        const float* __restrict__ g,
                        const float* __restrict__ b,
                        __bf16* __restrict__ hn) {
  __shared__ float s1[256];
  __shared__ float s2[256];
  const int row = blockIdx.x;
  const int tid = threadIdx.x;
  const float* xr = x + (size_t)row * D_DIM;
  const float* ar = attn + (size_t)row * D_DIM;

  float sum = 0.f, sq = 0.f;
  for (int j = tid; j < D_DIM; j += 256) {
    float h = xr[j] + ar[j];
    sum += h; sq += h * h;
  }
  s1[tid] = sum; s2[tid] = sq; __syncthreads();
  for (int st = 128; st > 0; st >>= 1) {
    if (tid < st) { s1[tid] += s1[tid + st]; s2[tid] += s2[tid + st]; }
    __syncthreads();
  }
  const float mu = s1[0] * (1.f / D_DIM);
  const float var = s2[0] * (1.f / D_DIM) - mu * mu;
  const float rstd = rsqrtf(var + LN_EPS);

  __bf16* hr = hn + (size_t)row * D_DIM;
  for (int j = tid; j < D_DIM; j += 256) {
    float h = xr[j] + ar[j];
    hr[j] = (__bf16)((h - mu) * rstd * g[j] + b[j]);
  }
}

// ---------------------------------------------------------------------------
// Host launcher
// ---------------------------------------------------------------------------
extern "C" void kernel_launch(void* const* d_in, const int* in_sizes, int n_in,
                              void* d_out, int out_size, void* d_ws, size_t ws_size,
                              hipStream_t stream) {
  (void)in_sizes; (void)n_in; (void)out_size; (void)ws_size;
  const int*   tok  = (const int*)  d_in[0];
  const float* Wemb = (const float*)d_in[1];
  const float* Wpos = (const float*)d_in[2];
  const float* Wq   = (const float*)d_in[3];
  const float* bq   = (const float*)d_in[4];
  const float* Wk   = (const float*)d_in[5];
  const float* bk   = (const float*)d_in[6];
  const float* Wv   = (const float*)d_in[7];
  const float* bv   = (const float*)d_in[8];
  const float* ln_g = (const float*)d_in[9];
  const float* ln_b = (const float*)d_in[10];
  const float* W1   = (const float*)d_in[11];
  const float* b1   = (const float*)d_in[12];
  const float* W2   = (const float*)d_in[13];
  const float* b2   = (const float*)d_in[14];
  float* logits = (float*)d_out;                       // [S, V]

  // ---- workspace carve (~442 MB total) ----
  size_t off = 0;
  auto carve = [&](size_t bytes) -> void* {
    void* p = (char*)d_ws + off;
    off += (bytes + 255) & ~(size_t)255;
    return p;
  };
  const size_t SD  = (size_t)S_LEN * D_DIM;
  const size_t SS  = (size_t)S_LEN * S_LEN;
  const size_t SDF = (size_t)S_LEN * DF_DIM;

  float*  x_f    = (float*) carve(SD * 4);
  __bf16* x_b    = (__bf16*)carve(SD * 2);
  __bf16* q_b    = (__bf16*)carve(SD * 2);
  __bf16* k_b    = (__bf16*)carve(SD * 2);
  __bf16* v_b    = (__bf16*)carve(SD * 2);
  float*  scores = (float*) carve(SS * 4);
  __bf16* probs  = (__bf16*)carve(SS * 2);
  __bf16* vT     = (__bf16*)carve((size_t)DH_DIM * S_LEN * 2);
  float*  attn   = (float*) carve(SD * 4);
  __bf16* hn_b   = (__bf16*)carve(SD * 2);
  __bf16* m_b    = (__bf16*)carve(SDF * 2);
  __bf16* o_b    = (__bf16*)carve(SD * 2);
  __bf16* Wq_b   = (__bf16*)carve((size_t)D_DIM * D_DIM * 2);
  __bf16* Wk_b   = (__bf16*)carve((size_t)D_DIM * D_DIM * 2);
  __bf16* Wv_b   = (__bf16*)carve((size_t)D_DIM * D_DIM * 2);
  __bf16* W1_b   = (__bf16*)carve((size_t)DF_DIM * D_DIM * 2);
  __bf16* W2_b   = (__bf16*)carve((size_t)D_DIM * DF_DIM * 2);
  __bf16* Wemb_b = (__bf16*)carve((size_t)NPAD_V * D_DIM * 2);

  auto blk = [](long n) { return (unsigned)((n + 255) / 256); };

  // ---- weight conversion fp32 -> bf16 ----
  cvt_f32_to_bf16<<<blk((long)D_DIM * D_DIM),  256, 0, stream>>>(Wq, Wq_b, (long)D_DIM * D_DIM);
  cvt_f32_to_bf16<<<blk((long)D_DIM * D_DIM),  256, 0, stream>>>(Wk, Wk_b, (long)D_DIM * D_DIM);
  cvt_f32_to_bf16<<<blk((long)D_DIM * D_DIM),  256, 0, stream>>>(Wv, Wv_b, (long)D_DIM * D_DIM);
  cvt_f32_to_bf16<<<blk((long)DF_DIM * D_DIM), 256, 0, stream>>>(W1, W1_b, (long)DF_DIM * D_DIM);
  cvt_f32_to_bf16<<<blk((long)D_DIM * DF_DIM), 256, 0, stream>>>(W2, W2_b, (long)D_DIM * DF_DIM);
  cvt_f32_to_bf16<<<blk((long)V_VOCAB * D_DIM), 256, 0, stream>>>(Wemb, Wemb_b, (long)V_VOCAB * D_DIM);
  fill_bf16_zero<<<blk((long)(NPAD_V - V_VOCAB) * D_DIM), 256, 0, stream>>>(
      Wemb_b + (size_t)V_VOCAB * D_DIM, (long)(NPAD_V - V_VOCAB) * D_DIM);

  // ---- embed + positional ----
  embed_kernel<<<blk((long)SD), 256, 0, stream>>>(tok, Wemb, Wpos, x_f, x_b);

  // ---- QKV projections: q = x @ Wq^T + bq (bf16 out) ----
  dim3 gSD(D_DIM / 128, S_LEN / 128);
  wmma_gemm_nt<<<gSD, 256, 0, stream>>>(x_b, D_DIM, Wq_b, D_DIM, bq, 1.f,
                                        nullptr, q_b, (long)D_DIM, D_DIM, D_DIM, 0);
  wmma_gemm_nt<<<gSD, 256, 0, stream>>>(x_b, D_DIM, Wk_b, D_DIM, bk, 1.f,
                                        nullptr, k_b, (long)D_DIM, D_DIM, D_DIM, 0);
  wmma_gemm_nt<<<gSD, 256, 0, stream>>>(x_b, D_DIM, Wv_b, D_DIM, bv, 1.f,
                                        nullptr, v_b, (long)D_DIM, D_DIM, D_DIM, 0);

  // ---- causal attention, head by head ----
  const float scale = 0.0625f;  // 1/sqrt(256)
  dim3 gScore(S_LEN / 128, S_LEN / 128);
  dim3 gAttn(DH_DIM / 128, S_LEN / 128);
  for (int h = 0; h < H_HEADS; ++h) {
    const __bf16* qh = q_b + (size_t)h * DH_DIM;
    const __bf16* kh = k_b + (size_t)h * DH_DIM;
    // scores = (q_h @ k_h^T) / 16   [S, S] fp32
    wmma_gemm_nt<<<gScore, 256, 0, stream>>>(qh, D_DIM, kh, D_DIM, nullptr, scale,
                                             scores, nullptr, (long)S_LEN,
                                             S_LEN, DH_DIM, 0);
    softmax_causal<<<S_LEN, 256, 0, stream>>>(scores, probs);
    transpose_head<<<blk((long)DH_DIM * S_LEN), 256, 0, stream>>>(v_b, vT, h);
    // attn_h = probs @ v_h  ->  fp32 into attn[:, h*DH : (h+1)*DH]
    wmma_gemm_nt<<<gAttn, 256, 0, stream>>>(probs, S_LEN, vT, S_LEN, nullptr, 1.f,
                                            attn + (size_t)h * DH_DIM, nullptr,
                                            (long)D_DIM, DH_DIM, S_LEN, 0);
  }

  // ---- residual + LayerNorm ----
  residual_layernorm<<<S_LEN, 256, 0, stream>>>(x_f, attn, ln_g, ln_b, hn_b);

  // ---- MLP: m = gelu(hn @ W1^T + b1), o = m @ W2^T + b2 ----
  dim3 gMlp1(DF_DIM / 128, S_LEN / 128);
  wmma_gemm_nt<<<gMlp1, 256, 0, stream>>>(hn_b, D_DIM, W1_b, D_DIM, b1, 1.f,
                                          nullptr, m_b, (long)DF_DIM,
                                          DF_DIM, D_DIM, 1 /*gelu*/);
  wmma_gemm_nt<<<gSD, 256, 0, stream>>>(m_b, DF_DIM, W2_b, DF_DIM, b2, 1.f,
                                        nullptr, o_b, (long)D_DIM, D_DIM, DF_DIM, 0);

  // ---- tied LM head: logits = o @ Wemb^T  [S, V] fp32 -> d_out ----
  dim3 gHead(NPAD_V / 128, S_LEN / 128);
  wmma_gemm_nt<<<gHead, 256, 0, stream>>>(o_b, D_DIM, Wemb_b, D_DIM, nullptr, 1.f,
                                          logits, nullptr, (long)V_VOCAB,
                                          V_VOCAB, D_DIM, 0);
}
